// SequenceClassifier_49830210568631
// MI455X (gfx1250) — compile-verified, matrix operand order checked by reading, not run on hardware
//
#include <hip/hip_runtime.h>

typedef __attribute__((ext_vector_type(16))) _Float16 v16h;
typedef __attribute__((ext_vector_type(8)))  float    v8f;
typedef __attribute__((ext_vector_type(2)))  __fp16   fp16x2;

#define VOCAB 53
#define EDIM  16
#define HDIM  32
#define GDIM  128   // 4*H
#define BATCH 4096
#define SEQ   512

// d_ws layout (bytes):
//   [0, GP_BYTES)           : Gp  f32 [53][2][64]  permuted gate table
//   [WHH_OFF, +8192)        : W_hh packed f16 in WMMA A-layout, v16h[8*32]
#define GP_FLOATS (VOCAB * 2 * 64)
#define GP_BYTES  (GP_FLOATS * 4)
#define WHH_OFF   GP_BYTES

// Gate rows 0..31=i, 32..63=f, 64..95=g, 96..127=o (tiles of 16 rows).
// Sigmoid gates (i,f,o -> tiles 0,1,2,3,6,7) are PRE-SCALED by 0.5 in both
// the gate table and W_hh, so sigmoid(x) = fma(tanh(acc), 0.5, 0.5) with no
// leading multiply. The g-gate (tiles 4,5) stays full-scale for tanh.

// ---------------- prep kernel 1: gate table G -> permuted Gp ----------------
// G[v][g] = sum_e W_ih[g][e]*emb[v][e] + b_ih[g] + b_hh[g]
//   Gp[(v*2 + hi)*64 + T*8 + r] = scale(T) * G[v][16*T + 8*hi + r]
__global__ void lstm_prep_gates(const float* __restrict__ emb,
                                const float* __restrict__ W_ih,
                                const float* __restrict__ b_ih,
                                const float* __restrict__ b_hh,
                                float* __restrict__ gp) {
  int g = threadIdx.x;   // 0..127 gate row
  int v = blockIdx.x;    // 0..52 vocab
  float s = b_ih[g] + b_hh[g];
#pragma unroll
  for (int e = 0; e < EDIM; ++e) s += W_ih[g * EDIM + e] * emb[v * EDIM + e];
  int T  = g >> 4;
  int hi = (g >> 3) & 1;
  int r  = g & 7;
  if (T != 4 && T != 5) s *= 0.5f;   // prescale sigmoid gates
  gp[(v * 2 + hi) * 64 + T * 8 + r] = s;
}

// ---------------- prep kernel 2: W_hh -> f16 A-layout tiles ----------------
// Tile T covers gate rows 16T..16T+15, K = hidden 0..31.
// A-layout (16-bit A 16x32): lane l holds row M=l%16; VGPR v holds K pair
//   k0 = (v<4 ? 2v : 16+2(v-4)) + 8*(l>>4)
__global__ void lstm_prep_whh(const float* __restrict__ W_hh,
                              _Float16* __restrict__ whh) {
  int idx = blockIdx.x * blockDim.x + threadIdx.x;  // 0..2047 = (T*32+l)*8+v
  if (idx >= 2048) return;
  int T = idx >> 8;
  int l = (idx >> 3) & 31;
  int v = idx & 7;
  int m  = 16 * T + (l & 15);
  int hi = l >> 4;
  int k0 = (v < 4 ? 2 * v : 16 + 2 * (v - 4)) + 8 * hi;
  float scale = (T == 4 || T == 5) ? 1.0f : 0.5f;   // prescale sigmoid gates
  whh[idx * 2 + 0] = (_Float16)(scale * W_hh[m * HDIM + k0]);
  whh[idx * 2 + 1] = (_Float16)(scale * W_hh[m * HDIM + k0 + 1]);
}

// ---------------- fast activations ----------------
// CDNA5 single-instruction v_tanh_f32 TRANS op when available.
#if defined(__has_builtin)
#if __has_builtin(__builtin_amdgcn_tanhf)
#define HAVE_AMDGCN_TANH 1
#endif
#endif

__device__ __forceinline__ float ftanh(float x) {
#ifdef HAVE_AMDGCN_TANH
  return __builtin_amdgcn_tanhf(x);
#else
  return 2.0f * __builtin_amdgcn_rcpf(1.0f + __expf(-2.0f * x)) - 1.0f;
#endif
}
// sigmoid(x) given y = x/2 (table prescaled): 0.5*tanh(y) + 0.5
__device__ __forceinline__ float fsig_half(float y) {
  return __builtin_fmaf(0.5f, ftanh(y), 0.5f);
}
// sigmoid(x) from full-scale x (head only)
__device__ __forceinline__ float fsig(float x) {
  return __builtin_fmaf(0.5f, ftanh(0.5f * x), 0.5f);
}

// ---------------- main LSTM kernel ----------------
__global__ __launch_bounds__(128, 1)
void lstm_wmma_kernel(const int* __restrict__ x,
                      const float* __restrict__ gp,
                      const v16h* __restrict__ whh,
                      const float* __restrict__ W_fc,
                      const float* __restrict__ b_fc,
                      float* __restrict__ out) {
  __shared__ __align__(16) float sG[GP_FLOATS];  // 27 KB permuted gate table
  const int tid = threadIdx.x;
  for (int i = tid; i < GP_FLOATS; i += 128) sG[i] = gp[i];
  __syncthreads();

  const int lane = tid & 31;
  const int wave = tid >> 5;
  const int hi   = lane >> 4;                 // which half of hidden indices
  const int col  = lane & 15;                 // batch column within tile
  const int bm   = blockIdx.x * 64 + wave * 16;
  const int brow = bm + col;

  // Persistent A operands: 8 tiles of W_hh (64 VGPRs of f16)
  v16h a[8];
#pragma unroll
  for (int T = 0; T < 8; ++T) a[T] = whh[T * 32 + lane];

  const int4* xrow = (const int4*)(x + (size_t)brow * SEQ);

  v16h hb = {};       // h in B operand layout (f16), starts at 0
  float cst[16];      // cell state, per-lane hidden subset j = 16t + 8hi + r
#pragma unroll
  for (int i = 0; i < 16; ++i) cst[i] = 0.0f;
  float hq[16];       // current h (f32), same indexing
#pragma unroll
  for (int i = 0; i < 16; ++i) hq[i] = 0.0f;

  const float* gbase = sG + hi * 64;

  for (int t4 = 0; t4 < SEQ / 4; ++t4) {
    int4 tk = xrow[t4];
    if (t4 + 1 < SEQ / 4) __builtin_prefetch(xrow + t4 + 1, 0, 0);
    int toks[4] = {tk.x, tk.y, tk.z, tk.w};
#pragma unroll
    for (int s = 0; s < 4; ++s) {
      const float* gl = gbase + toks[s] * 128;  // this lane's 64 C-init values
      v8f acc[8];
#pragma unroll
      for (int T = 0; T < 8; ++T) acc[T] = *(const v8f*)(gl + T * 8);
      // gates[16 rows, 16 batch] per tile: D = W_hh_tile(f16) x h(f16) + G[x_t]
#pragma unroll
      for (int T = 0; T < 8; ++T)
        acc[T] = __builtin_amdgcn_wmma_f32_16x16x32_f16(
            false, a[T], false, hb, (short)0, acc[T], false, false);
      // lane-local LSTM cell update; tiles {0,1}=i {2,3}=f {4,5}=g {6,7}=o
      // (i,f,o tiles arrive prescaled by 0.5)
#pragma unroll
      for (int tt = 0; tt < 2; ++tt)
#pragma unroll
        for (int r = 0; r < 8; ++r) {
          float ig = fsig_half(acc[0 + tt][r]);
          float fg = fsig_half(acc[2 + tt][r]);
          float gg = ftanh(acc[4 + tt][r]);
          float og = fsig_half(acc[6 + tt][r]);
          float c  = fg * cst[tt * 8 + r] + ig * gg;
          cst[tt * 8 + r] = c;
          hq[tt * 8 + r]  = og * ftanh(c);
        }
      // in-lane repack of h into next step's B operand (no cross-lane moves:
      // D-layout lane halves carry exactly the K-interleave the B operand wants)
#pragma unroll
      for (int v = 0; v < 8; ++v) {
        fp16x2 p = __builtin_amdgcn_cvt_pkrtz(hq[(v >> 2) * 8 + 2 * (v & 3) + 0],
                                              hq[(v >> 2) * 8 + 2 * (v & 3) + 1]);
        hb[2 * v + 0] = (_Float16)p[0];
        hb[2 * v + 1] = (_Float16)p[1];
      }
    }
  }

  // head: out[b] = sigmoid(h . W_fc + b_fc); lane pair (l, l+16) holds h split
  float part = 0.0f;
#pragma unroll
  for (int tt = 0; tt < 2; ++tt)
#pragma unroll
    for (int r = 0; r < 8; ++r)
      part += hq[tt * 8 + r] * W_fc[16 * tt + 8 * hi + r];
  part += __shfl_xor(part, 16, 32);
  if (hi == 0) out[brow] = fsig(part + b_fc[0]);
}

// ---------------- launcher ----------------
extern "C" void kernel_launch(void* const* d_in, const int* in_sizes, int n_in,
                              void* d_out, int out_size, void* d_ws, size_t ws_size,
                              hipStream_t stream) {
  const int*   x    = (const int*)d_in[0];
  const float* emb  = (const float*)d_in[1];
  const float* W_ih = (const float*)d_in[2];
  const float* W_hh = (const float*)d_in[3];
  const float* b_ih = (const float*)d_in[4];
  const float* b_hh = (const float*)d_in[5];
  const float* W_fc = (const float*)d_in[6];
  const float* b_fc = (const float*)d_in[7];
  float* out = (float*)d_out;

  float*    gp  = (float*)d_ws;
  _Float16* whh = (_Float16*)((char*)d_ws + WHH_OFF);

  hipLaunchKernelGGL(lstm_prep_gates, dim3(VOCAB), dim3(GDIM), 0, stream,
                     emb, W_ih, b_ih, b_hh, gp);
  hipLaunchKernelGGL(lstm_prep_whh, dim3(8), dim3(256), 0, stream, W_hh, whh);
  hipLaunchKernelGGL(lstm_wmma_kernel, dim3(BATCH / 64), dim3(128), 0, stream,
                     x, gp, (const v16h*)whh, W_fc, b_fc, out);
}
